// MultiHeadSelfAttention_26611617366407
// MI455X (gfx1250) — matrix operand order repaired and checked by reading, not compiled, not run on hardware
//
#include <hip/hip_runtime.h>

// MI455X / gfx1250 fused multi-head self-attention forward.
// f16 WMMA (v_wmma_f32_16x16x32_f16) + TDM (tensor_load_to_lds) staging.
// Intermediates (Q,K,Vt,attn-out) kept as f16 in workspace (64MB < 192MB L2).

typedef _Float16 half_t;
typedef __attribute__((ext_vector_type(16))) _Float16     v16h;
typedef __attribute__((ext_vector_type(8)))  float        v8f;
typedef __attribute__((ext_vector_type(4)))  float        f32x4;
typedef __attribute__((ext_vector_type(4)))  unsigned int u32x4;
typedef __attribute__((ext_vector_type(2)))  unsigned int u32x2;
typedef __attribute__((ext_vector_type(8)))  int          i32x8;
typedef __attribute__((ext_vector_type(4)))  int          i32x4;

union Frag {
    v16h  v;
    u32x4 q[2];
};

#define BATCH 4
#define SEQ   2048
#define EMB   1024
#define NH    16
#define HD    64

static __device__ __forceinline__ v8f wmma_f16(const Frag& a, const Frag& b, v8f c) {
    return __builtin_amdgcn_wmma_f32_16x16x32_f16(
        false, a.v, false, b.v, (short)0, c, false, false);
}

// ---------------------------------------------------------------------------
// TDM: 2D tile load Global -> LDS, data_size = 8B units. D# per ISA 8.3/8.4.
//   tile_d0u : tile width in 8-byte units   (tile_dim0, tensor_dim0)
//   tile_d1  : number of rows               (tile_dim1, tensor_dim1)
//   s0u      : row stride in 8-byte units   (tensor_dim0_stride)
// ---------------------------------------------------------------------------
static __device__ __forceinline__ void tdm_load_2d(unsigned lds_off, const void* gptr,
                                                   unsigned tile_d0u, unsigned tile_d1,
                                                   unsigned s0u)
{
    unsigned long long ga = (unsigned long long)gptr;
    u32x4 g0;
    g0[0] = 1u;                                   // count=1, user descriptor
    g0[1] = lds_off;                              // lds_addr (bytes)
    g0[2] = (unsigned)ga;                         // global_addr[31:0]
    g0[3] = (unsigned)((ga >> 32) & 0x1FFFFFFu)   // global_addr[56:32]
            | (2u << 30);                         // type = 2 ("image")
    i32x8 g1;
    g1[0] = (int)(3u << 16);                                      // data_size=8B, mask=0
    g1[1] = (int)((tile_d0u & 0xFFFFu) << 16);                    // tensor_dim0[15:0]
    g1[2] = (int)(((tile_d0u >> 16) & 0xFFFFu)                    // tensor_dim0[31:16]
            | ((tile_d1 & 0xFFFFu) << 16));                       // tensor_dim1[15:0]
    g1[3] = (int)(((tile_d1 >> 16) & 0xFFFFu)                     // tensor_dim1[31:16]
            | ((tile_d0u & 0xFFFFu) << 16));                      // tile_dim0
    g1[4] = (int)(tile_d1 & 0xFFFFu);                             // tile_dim1, tile_dim2=0
    g1[5] = (int)s0u;                                             // dim0_stride[31:0]
    g1[6] = 0;                                                    // dim0_stride[47:32], dim1_stride lo
    g1[7] = 0;
    i32x4 gz = {0, 0, 0, 0};
#if defined(__clang_major__) && (__clang_major__ >= 23)
    i32x8 gz8 = {0, 0, 0, 0, 0, 0, 0, 0};
    __builtin_amdgcn_tensor_load_to_lds(g0, g1, gz, gz, gz8, 0);
#else
    __builtin_amdgcn_tensor_load_to_lds(g0, g1, gz, gz, 0);
#endif
}

// ---------------------------------------------------------------------------
// Generic f16-WMMA GEMM: C[M,N] = A[M,K] * B[K,N]   (unchanged from round 1)
// ---------------------------------------------------------------------------
template<bool A_HALF, bool QKV_SCATTER>
__global__ __launch_bounds__(256)
void gemm_wmma_kernel(const float* __restrict__ Af, const half_t* __restrict__ Ah,
                      const float* __restrict__ Bw,
                      int M, int N, int K,
                      float* __restrict__ Cout,
                      half_t* __restrict__ Qh, half_t* __restrict__ Kh,
                      half_t* __restrict__ Vt)
{
    __shared__ half_t As[128][40];
    __shared__ half_t Bt[128][40];

    const int tid  = threadIdx.x;
    const int lane = tid & 31, sub = lane >> 4, l15 = lane & 15;
    const int wave = tid >> 5;
    const int wm = (wave >> 2) * 64;
    const int wn = (wave & 3) * 32;
    const int tileM = blockIdx.y * 128;
    const int tileN = blockIdx.x * 128;

    v8f zero = {0.f,0.f,0.f,0.f,0.f,0.f,0.f,0.f};
    v8f acc[4][2];
    #pragma unroll
    for (int i = 0; i < 4; ++i)
        #pragma unroll
        for (int j = 0; j < 2; ++j) acc[i][j] = zero;

    for (int kk = 0; kk < K; kk += 32) {
        __syncthreads();
        #pragma unroll
        for (int i = 0; i < 4; ++i) {
            int c    = tid + i * 256;
            int row  = c >> 3;
            int col4 = (c & 7) << 2;
            if (A_HALF) {
                const half_t* src = Ah + (size_t)(tileM + row) * K + kk + col4;
                *(u32x2*)&As[row][col4] = *(const u32x2*)src;
            } else {
                const float* src = Af + (size_t)(tileM + row) * K + kk + col4;
                f32x4 f = *(const f32x4*)src;
                As[row][col4+0] = (half_t)f.x;
                As[row][col4+1] = (half_t)f.y;
                As[row][col4+2] = (half_t)f.z;
                As[row][col4+3] = (half_t)f.w;
            }
        }
        #pragma unroll
        for (int i = 0; i < 4; ++i) {
            int c    = tid + i * 256;
            int krow = c >> 5;
            int n4   = (c & 31) << 2;
            const float* src = Bw + (size_t)(kk + krow) * N + tileN + n4;
            f32x4 f = *(const f32x4*)src;
            Bt[n4+0][krow] = (half_t)f.x;
            Bt[n4+1][krow] = (half_t)f.y;
            Bt[n4+2][krow] = (half_t)f.z;
            Bt[n4+3][krow] = (half_t)f.w;
        }
        __syncthreads();

        Frag a[4], b[2];
        #pragma unroll
        for (int mi = 0; mi < 4; ++mi) {
            const half_t* r = &As[wm + mi*16 + l15][0];
            a[mi].q[0] = *(const u32x4*)(r + 8*sub);
            a[mi].q[1] = *(const u32x4*)(r + 16 + 8*sub);
        }
        #pragma unroll
        for (int ni = 0; ni < 2; ++ni) {
            const half_t* r = &Bt[wn + ni*16 + l15][0];
            b[ni].q[0] = *(const u32x4*)(r + 8*sub);
            b[ni].q[1] = *(const u32x4*)(r + 16 + 8*sub);
        }
        #pragma unroll
        for (int mi = 0; mi < 4; ++mi)
            #pragma unroll
            for (int ni = 0; ni < 2; ++ni)
                acc[mi][ni] = wmma_f16(a[mi], b[ni], acc[mi][ni]);
    }

    #pragma unroll
    for (int mi = 0; mi < 4; ++mi)
        #pragma unroll
        for (int ni = 0; ni < 2; ++ni)
            #pragma unroll
            for (int r = 0; r < 8; ++r) {
                int gm = tileM + wm + mi*16 + r + 8*sub;
                int gn = tileN + wn + ni*16 + l15;
                float v = acc[mi][ni][r];
                if (QKV_SCATTER) {
                    int bb = gm >> 11, t = gm & (SEQ-1);
                    int part = gn >> 10, cc = gn & (EMB-1);
                    int hh = cc >> 6, d = cc & (HD-1);
                    size_t hidx = ((size_t)(bb*NH + hh) * SEQ + t) * HD + d;
                    if (part == 0)      Qh[hidx] = (half_t)(v * 0.125f); // 1/sqrt(64)
                    else if (part == 1) Kh[hidx] = (half_t)v;
                    else Vt[((size_t)(bb*NH + hh) * HD + d) * SEQ + t] = (half_t)v;
                } else {
                    Cout[(size_t)gm * N + gn] = v;
                }
            }
}

// ---------------------------------------------------------------------------
// Flash attention: block = 4 waves x 32 queries = 128 queries of one (b,h).
// K/V 64-key tiles staged to LDS by TDM (double-buffered, wave 0 issues,
// s_wait_tensorcnt + block barrier publishes). Online softmax, all matrix
// work in v_wmma_f32_16x16x32_f16.
// ---------------------------------------------------------------------------
__global__ __launch_bounds__(128)
void flash_attn_kernel(const half_t* __restrict__ Qh, const half_t* __restrict__ Kh,
                       const half_t* __restrict__ Vt, half_t* __restrict__ Oh)
{
    __shared__ half_t Ktile[2][64][64];   // [buf][key][d]      8KB each
    __shared__ half_t Vtile[2][64][64];   // [buf][d][key]      8KB each
    __shared__ half_t Pw[4][32][64];      // per-wave P relay   16KB

    const int tid  = threadIdx.x;
    const int lane = tid & 31, sub = lane >> 4, l15 = lane & 15;
    const int wave = tid >> 5;

    const int bid = blockIdx.x;
    const int qt  = bid & (SEQ/128 - 1);  // 16 q-blocks of 128
    const int bh  = bid >> 4;             // 0..63
    const int h   = bh & (NH-1), b = bh >> 4;
    const int q0b = qt * 128;             // block's first query
    const int q0w = q0b + wave * 32;      // wave's first query

    const half_t* Qb = Qh + (size_t)bh * SEQ * HD;
    const half_t* Kb = Kh + (size_t)bh * SEQ * HD;
    const half_t* Vb = Vt + (size_t)bh * HD * SEQ;   // [D][T]

    // Q A-fragments: 2 row-tiles x 2 D-chunks, held in registers all loop
    Frag qa[2][2];
    #pragma unroll
    for (int m = 0; m < 2; ++m)
        #pragma unroll
        for (int c = 0; c < 2; ++c) {
            const half_t* qr = Qb + (size_t)(q0w + m*16 + l15) * HD + 32*c;
            qa[m][c].q[0] = *(const u32x4*)(qr + 8*sub);
            qa[m][c].q[1] = *(const u32x4*)(qr + 16 + 8*sub);
        }

    v8f zero = {0.f,0.f,0.f,0.f,0.f,0.f,0.f,0.f};
    v8f o[2][4];
    #pragma unroll
    for (int m = 0; m < 2; ++m)
        #pragma unroll
        for (int dj = 0; dj < 4; ++dj) o[m][dj] = zero;
    float m_i[2][8], l_i[2][8];
    #pragma unroll
    for (int m = 0; m < 2; ++m)
        #pragma unroll
        for (int r = 0; r < 8; ++r) { m_i[m][r] = -1e30f; l_i[m][r] = 0.f; }

    const int kend = q0b + 128;           // causal: block needs keys < q0b+128

    // Prologue: wave 0 issues TDM loads for tile 0 into buffer 0
    if (wave == 0) {
        tdm_load_2d((unsigned)(size_t)&Ktile[0][0][0], Kb, /*8KB contig*/1024, 1, 1024);
        tdm_load_2d((unsigned)(size_t)&Vtile[0][0][0], Vb, 16, 64, SEQ*2/8);
    }

    int ib = 0;
    for (int kt = 0; kt < kend; kt += 64, ib ^= 1) {
        // Issue next tile; wait for current tile's pair to land
        if (wave == 0) {
            if (kt + 64 < kend) {
                tdm_load_2d((unsigned)(size_t)&Ktile[ib^1][0][0],
                            Kb + (size_t)(kt + 64) * HD, 1024, 1, 1024);
                tdm_load_2d((unsigned)(size_t)&Vtile[ib^1][0][0],
                            Vb + (kt + 64), 16, 64, SEQ*2/8);
                __builtin_amdgcn_s_wait_tensorcnt(2);
            } else {
                __builtin_amdgcn_s_wait_tensorcnt(0);
            }
        }
        __syncthreads();   // publish Ktile/Vtile[ib] to all waves

        // ---- S = Q K^T : 2 row-tiles x 4 key-frags x 2 D-chunks = 16 WMMAs
        v8f s[2][4];
        #pragma unroll
        for (int m = 0; m < 2; ++m)
            #pragma unroll
            for (int kf = 0; kf < 4; ++kf) s[m][kf] = zero;
        #pragma unroll
        for (int kf = 0; kf < 4; ++kf)
            #pragma unroll
            for (int c = 0; c < 2; ++c) {
                Frag kb;
                const half_t* kr = &Ktile[ib][kf*16 + l15][32*c];
                kb.q[0] = *(const u32x4*)(kr + 8*sub);
                kb.q[1] = *(const u32x4*)(kr + 16 + 8*sub);
                #pragma unroll
                for (int m = 0; m < 2; ++m)
                    s[m][kf] = wmma_f16(qa[m][c], kb, s[m][kf]);
            }

        // ---- causal mask + online softmax ----
        #pragma unroll
        for (int m = 0; m < 2; ++m)
            #pragma unroll
            for (int r = 0; r < 8; ++r) {
                int qrow = q0w + m*16 + r + 8*sub;
                float sv[4];
                #pragma unroll
                for (int kf = 0; kf < 4; ++kf) {
                    sv[kf] = s[m][kf][r];
                    if (kt + kf*16 + l15 > qrow) sv[kf] = -1e30f;
                }
                float mloc = fmaxf(fmaxf(sv[0], sv[1]), fmaxf(sv[2], sv[3]));
                #pragma unroll
                for (int off = 1; off < 16; off <<= 1)
                    mloc = fmaxf(mloc, __shfl_xor(mloc, off, 32));
                float mnew  = fmaxf(m_i[m][r], mloc);
                float alpha = __expf(m_i[m][r] - mnew);
                float p[4], rs = 0.f;
                #pragma unroll
                for (int kf = 0; kf < 4; ++kf) { p[kf] = __expf(sv[kf] - mnew); rs += p[kf]; }
                #pragma unroll
                for (int off = 1; off < 16; off <<= 1)
                    rs += __shfl_xor(rs, off, 32);
                l_i[m][r] = l_i[m][r] * alpha + rs;
                m_i[m][r] = mnew;
                #pragma unroll
                for (int dj = 0; dj < 4; ++dj) o[m][dj][r] = o[m][dj][r] * alpha;
                #pragma unroll
                for (int kf = 0; kf < 4; ++kf)
                    Pw[wave][m*16 + r + 8*sub][kf*16 + l15] = (half_t)p[kf];
            }
        __syncthreads();   // P relay visible (and waves in lockstep on buffers)

        // ---- O += P V : 2 row-tiles x 4 D-frags x 2 key-chunks = 16 WMMAs
        Frag pa[2][2];
        #pragma unroll
        for (int m = 0; m < 2; ++m)
            #pragma unroll
            for (int kc = 0; kc < 2; ++kc) {
                const half_t* pr = &Pw[wave][m*16 + l15][kc*32];
                pa[m][kc].q[0] = *(const u32x4*)(pr + 8*sub);
                pa[m][kc].q[1] = *(const u32x4*)(pr + 16 + 8*sub);
            }
        #pragma unroll
        for (int dj = 0; dj < 4; ++dj)
            #pragma unroll
            for (int kc = 0; kc < 2; ++kc) {
                Frag vb;
                const half_t* vr = &Vtile[ib][dj*16 + l15][kc*32];
                vb.q[0] = *(const u32x4*)(vr + 8*sub);
                vb.q[1] = *(const u32x4*)(vr + 16 + 8*sub);
                #pragma unroll
                for (int m = 0; m < 2; ++m)
                    o[m][dj] = wmma_f16(pa[m][kc], vb, o[m][dj]);
            }
        __syncthreads();   // all reads of buffer ib done before it is reloaded
    }

    // ---- normalize + store f16 [B,T,C] ----
    #pragma unroll
    for (int m = 0; m < 2; ++m) {
        size_t obase = ((size_t)b * SEQ + q0w + m*16) * EMB + h * HD;
        #pragma unroll
        for (int dj = 0; dj < 4; ++dj)
            #pragma unroll
            for (int r = 0; r < 8; ++r) {
                float val = o[m][dj][r] / l_i[m][r];
                Oh[obase + (size_t)(r + 8*sub) * EMB + dj*16 + l15] = (half_t)val;
            }
    }
}

// ---------------------------------------------------------------------------
extern "C" void kernel_launch(void* const* d_in, const int* in_sizes, int n_in,
                              void* d_out, int out_size, void* d_ws, size_t ws_size,
                              hipStream_t stream)
{
    const float* x     = (const float*)d_in[0];   // [4,2048,1024]
    const float* w_qkv = (const float*)d_in[1];   // [1024,3072]
    const float* w_out = (const float*)d_in[2];   // [1024,1024]
    float* out = (float*)d_out;                   // [4,2048,1024]

    const size_t elems = (size_t)BATCH * SEQ * EMB;
    half_t* Qh = (half_t*)d_ws;
    half_t* Kh = Qh + elems;
    half_t* Vt = Kh + elems;
    half_t* Oh = Vt + elems;                      // 64MB of ws total

    dim3 blk(256);
    gemm_wmma_kernel<false, true><<<dim3(3*EMB/128, (BATCH*SEQ)/128), blk, 0, stream>>>(
        x, nullptr, w_qkv, BATCH*SEQ, 3*EMB, EMB, nullptr, Qh, Kh, Vt);
    flash_attn_kernel<<<dim3(BATCH*NH*(SEQ/128)), dim3(128), 0, stream>>>(Qh, Kh, Vt, Oh);
    gemm_wmma_kernel<true, false><<<dim3(EMB/128, (BATCH*SEQ)/128), blk, 0, stream>>>(
        nullptr, Oh, w_out, BATCH*SEQ, EMB, EMB, out, nullptr, nullptr, nullptr);
}
